// LSTM_44933947851218
// MI455X (gfx1250) — matrix operand order repaired and checked by reading, not compiled, not run on hardware
//
#include <hip/hip_runtime.h>
#include <cstdint>
#include <cstddef>

// ---------------- problem constants ----------------
#define B_   32
#define T_   1024
#define I_   1024
#define H_   1024
#define FH_  4096          // 4*H
#define L_   2
#define NBLK 64            // H/16 column-owner blocks
#define THREADS 512        // 16 wave32: waves 0-7 -> layer 0, waves 8-15 -> layer 1

typedef __attribute__((ext_vector_type(16))) __bf16 v16bf;
typedef __attribute__((ext_vector_type(8)))  float  v8f;
typedef unsigned int u32x4 __attribute__((ext_vector_type(4)));

// ---------------- workspace layout (bytes) ----------------
#define WS_BAR   0ull                               // {count,gen} + pad
#define WS_W     1024ull                            // swizzled bf16 weights
#define WS_W_SZ  (2ull*2*256*32*1024)               // 33,554,432  (L*mat*jt*kc*1KB)
#define WS_X     (WS_W + WS_W_SZ)                   // swizzled bf16 x
#define WS_X_SZ  (1024ull*2*32*1024)                // 67,108,864  (t*mt*kc*1KB)
#define WS_H     (WS_X + WS_X_SZ)                   // bf16 h state, double buffered
#define WS_H_SZ  (2ull*2*2*32*1024)                 // 262,144     (l*par*mt*kc*1KB)
#define WS_C     (WS_H + WS_H_SZ)                   // f32 c state [L][B][H]

__device__ __forceinline__ unsigned short f2bf(float f) {
  unsigned u = __float_as_uint(f);
  u += 0x7FFFu + ((u >> 16) & 1u);                  // round-to-nearest-even
  return (unsigned short)(u >> 16);
}

// A-operand (16x32 bf16) swizzle: element (m in [0,16), ko in [0,32)) -> byte in 1KB chunk
__device__ __forceinline__ size_t a_swz(unsigned m, unsigned ko) {
  unsigned lane = m + (((ko >> 3) & 1u) << 4);      // lanes 16-31 hold K 8..15 / 24..31
  return (size_t)lane * 32 + (ko >> 4) * 16 + (ko & 7u) * 2;
}

// ---------------- init: barrier, h0 -> bf16 swizzle, c0 -> f32 state ----------------
__global__ void k_init(const float* __restrict__ h0, const float* __restrict__ c0,
                       char* __restrict__ ws) {
  unsigned e = blockIdx.x * blockDim.x + threadIdx.x;           // 65536 threads
  if (e == 0) { *(unsigned*)(ws + WS_BAR) = 0u; *(unsigned*)(ws + WS_BAR + 4) = 0u; }
  if (e >= (unsigned)(L_ * B_ * H_)) return;
  unsigned l = e >> 15, b = (e >> 10) & 31u, h = e & 1023u;
  *(float*)(ws + WS_C + (size_t)e * 4) = c0[e];
  unsigned par = l;                                 // l0 read at phase0(par0), l1 at phase1(par1)
  size_t off = WS_H + ((((size_t)(l * 2 + par) * 2 + (b >> 4)) * 32 + (h >> 5)) << 10)
             + a_swz(b & 15u, h & 31u);
  *(unsigned short*)(ws + off) = f2bf(h0[e]);
}

// ---------------- weights f32 -> bf16, swizzled to WMMA B-operand layout ----------------
__global__ void k_wswz(const float* __restrict__ Wi, const float* __restrict__ Wh,
                       char* __restrict__ ws) {
  size_t e = (size_t)blockIdx.x * blockDim.x + threadIdx.x;     // 16,777,216
  unsigned mat = (unsigned)(e >> 23);
  unsigned l   = (unsigned)((e >> 22) & 1u);
  unsigned j   = (unsigned)((e >> 10) & 4095u);
  unsigned k   = (unsigned)(e & 1023u);
  float f = (mat == 0 ? Wi : Wh)[((size_t)l * FH_ + j) * I_ + k];
  unsigned ko = k & 31u;
  unsigned lane = (j & 15u) + (((ko >> 3) & 1u) << 4); // B: lanes 16-31 hold K 8..15 / 24..31
  size_t off = WS_W + ((((size_t)(l * 2 + mat) * 256 + (j >> 4)) * 32 + (k >> 5)) << 10)
             + (size_t)lane * 32 + (ko >> 4) * 16 + (ko & 7u) * 2;
  *(unsigned short*)(ws + off) = f2bf(f);
}

// ---------------- x f32 -> bf16, swizzled to WMMA A-operand layout ----------------
__global__ void k_xswz(const float* __restrict__ x, char* __restrict__ ws) {
  size_t e = (size_t)blockIdx.x * blockDim.x + threadIdx.x;     // 33,554,432
  unsigned b = (unsigned)(e >> 20), t = (unsigned)((e >> 10) & 1023u), i = (unsigned)(e & 1023u);
  size_t off = WS_X + ((((size_t)t * 2 + (b >> 4)) * 32 + (i >> 5)) << 10)
             + a_swz(b & 15u, i & 31u);
  *(unsigned short*)(ws + off) = f2bf(x[e]);
}

// ---------------- persistent pipelined LSTM ----------------
__global__ void __launch_bounds__(THREADS, 1)
k_lstm(char* __restrict__ ws, const float* __restrict__ bi, const float* __restrict__ bh,
       float* __restrict__ outp) {
  __shared__ float zbuf[L_][2][4][16][16];          // [layer][mt][gate][row][col] = 16 KB

  const int tid  = threadIdx.x;
  const int lane = tid & 31;
  const int wv   = tid >> 5;                        // 0..15
  const int lw   = wv >> 3;                         // this wave's layer (0 or 1)
  const int gate = wv & 3;
  const int mt   = (wv >> 2) & 1;
  const int nblk = blockIdx.x;                      // owns hidden cols [16*nblk, 16*nblk+16)
  const int jt   = gate * 64 + nblk;                // z-column tile index in [0,256)

  unsigned* cnt = (unsigned*)(ws + WS_BAR);
  unsigned* gen = (unsigned*)(ws + WS_BAR + 4);
  volatile unsigned* vgen = (volatile unsigned*)gen;

  float* cF  = (float*)(ws + WS_C);
  float* hxp = outp + (size_t)B_ * T_ * H_;
  float* cxp = hxp + (size_t)L_ * B_ * H_;
  float* igp = cxp + (size_t)L_ * B_ * H_;
  float* fgp = igp + (size_t)L_ * B_ * H_;
  float* cgp = fgp + (size_t)L_ * B_ * H_;
  float* ogp = cgp + (size_t)L_ * B_ * H_;

  // time-invariant per-wave values, hoisted off the critical path
  const int   j    = jt * 16 + (lane & 15);
  const float bsum = bi[lw * FH_ + j] + bh[lw * FH_ + j];
  const size_t wib = WS_W + ((((size_t)(lw * 2 + 0) * 256 + jt) * 32) << 10);
  const size_t whb = WS_W + ((((size_t)(lw * 2 + 1) * 256 + jt) * 32) << 10);
  const int   rowb = (lane >> 4) << 3;

  unsigned target = 1;
  for (int p = 0; p <= T_; ++p) {
    const int par = p & 1, wpar = par ^ 1;

    // ---------- GEMM tiles: waves 0-7 do layer 0 (t=p), waves 8-15 layer 1 (t=p-1) ----------
    {
      int t = p - lw;
      if (t >= 0 && t < T_) {
        size_t ain = (lw == 0)
          ? WS_X + ((((size_t)t * 2 + mt) * 32) << 10)                  // x[t]
          : WS_H + ((((size_t)(0 * 2 + par) * 2 + mt) * 32) << 10);     // h of layer 0, this step
        size_t arec = WS_H + ((((size_t)(lw * 2 + par) * 2 + mt) * 32) << 10);

        v8f acc = {0.f, 0.f, 0.f, 0.f, 0.f, 0.f, 0.f, 0.f};
        const char* pa = ws + ain + (size_t)lane * 32;
        const char* pb = ws + wib + (size_t)lane * 32;
        #pragma unroll 4
        for (int kc = 0; kc < 32; ++kc) {           // input-to-hidden, K = 1024
          union { u32x4 q[2]; v16bf v; } A, Bm;
          A.q[0]  = *(const u32x4*)(pa);       A.q[1]  = *(const u32x4*)(pa + 16);
          Bm.q[0] = *(const u32x4*)(pb);       Bm.q[1] = *(const u32x4*)(pb + 16);
          acc = __builtin_amdgcn_wmma_f32_16x16x32_bf16(false, A.v, false, Bm.v,
                                                        (short)0, acc, false, false);
          pa += 1024; pb += 1024;
        }
        pa = ws + arec + (size_t)lane * 32;
        pb = ws + whb + (size_t)lane * 32;
        #pragma unroll 4
        for (int kc = 0; kc < 32; ++kc) {           // hidden-to-hidden, K = 1024
          union { u32x4 q[2]; v16bf v; } A, Bm;
          A.q[0]  = *(const u32x4*)(pa);       A.q[1]  = *(const u32x4*)(pa + 16);
          Bm.q[0] = *(const u32x4*)(pb);       Bm.q[1] = *(const u32x4*)(pb + 16);
          acc = __builtin_amdgcn_wmma_f32_16x16x32_bf16(false, A.v, false, Bm.v,
                                                        (short)0, acc, false, false);
          pa += 1024; pb += 1024;
        }
        // bias + stage z tile to LDS (C layout: VGPR r -> row r (+8 for lanes 16-31))
        #pragma unroll
        for (int r = 0; r < 8; ++r)
          zbuf[lw][mt][gate][rowb + r][lane & 15] = acc[r] + bsum;
      }
    }
    __syncthreads();

    // ---------- block-local gate combine: c/h update, outputs (1 elem/thread/layer) ----------
    for (int l = 0; l < L_; ++l) {
      int t = p - l;
      if (t < 0 || t >= T_) continue;
      int e = tid;                                  // 512 = 32 rows x 16 cols
      int b = e >> 4, col = e & 15;
      float iv = zbuf[l][b >> 4][0][b & 15][col];
      float fv = zbuf[l][b >> 4][1][b & 15][col];
      float gv = zbuf[l][b >> 4][2][b & 15][col];
      float ov = zbuf[l][b >> 4][3][b & 15][col];
      iv = 1.f / (1.f + __expf(-iv));
      fv = 1.f / (1.f + __expf(-fv));
      gv = tanhf(gv);
      ov = 1.f / (1.f + __expf(-ov));
      int h = nblk * 16 + col;
      size_t ci = ((size_t)l * B_ + b) * H_ + h;
      float cold = cF[ci];
      float cnew = fv * cold + iv * gv;
      float hnew = ov * tanhf(cnew);
      cF[ci] = cnew;
      // write h (bf16, A-swizzled) into next-parity buffer
      size_t hoff = WS_H + ((((size_t)(l * 2 + wpar) * 2 + (b >> 4)) * 32 + (h >> 5)) << 10)
                  + a_swz(b & 15u, h & 31u);
      *(unsigned short*)(ws + hoff) = f2bf(hnew);
      if (l == 1) outp[((size_t)b * T_ + t) * H_ + h] = hnew;
      if (t == T_ - 1) {
        hxp[ci] = hnew; cxp[ci] = cnew;
        igp[ci] = iv; fgp[ci] = fv; cgp[ci] = gv; ogp[ci] = ov;
      }
    }

    // ---------- device-wide barrier (one per phase) ----------
    __syncthreads();
    if (tid == 0) {
      __threadfence();
      unsigned a = atomicAdd(cnt, 1u);
      if (a == (unsigned)(NBLK - 1)) { *cnt = 0u; __threadfence(); atomicAdd(gen, 1u); }
      else { while (*vgen < target) __builtin_amdgcn_s_sleep(2); }
    }
    __syncthreads();
    ++target;
  }
}

// ---------------- entry ----------------
extern "C" void kernel_launch(void* const* d_in, const int* in_sizes, int n_in,
                              void* d_out, int out_size, void* d_ws, size_t ws_size,
                              hipStream_t stream) {
  const float* x  = (const float*)d_in[0];
  const float* h0 = (const float*)d_in[1];
  const float* c0 = (const float*)d_in[2];
  const float* Wi = (const float*)d_in[3];
  const float* bi = (const float*)d_in[4];
  const float* Wh = (const float*)d_in[5];
  const float* bh = (const float*)d_in[6];
  char*  ws  = (char*)d_ws;
  float* out = (float*)d_out;

  k_init<<<256, 256, 0, stream>>>(h0, c0, ws);
  k_wswz<<<(2u * 2u * 4096u * 1024u) / 256u, 256, 0, stream>>>(Wi, Wh, ws);   // 65536 blocks
  k_xswz<<<(32u * 1024u * 1024u) / 256u, 256, 0, stream>>>(x, ws);            // 131072 blocks
  k_lstm<<<NBLK, THREADS, 0, stream>>>(ws, bi, bh, out);
}
